// CAModel_13769665151261
// MI455X (gfx1250) — matrix operand order repaired and checked by reading, not compiled, not run on hardware
//
#include <hip/hip_runtime.h>
#include <hip/hip_bf16.h>

typedef __attribute__((ext_vector_type(16))) __bf16   v16bf;
typedef __attribute__((ext_vector_type(8)))  __bf16   v8bf;
typedef __attribute__((ext_vector_type(8)))  float    v8f;
typedef __attribute__((ext_vector_type(4)))  float    v4f;
typedef __attribute__((ext_vector_type(8)))  unsigned v8u;

static constexpr int Bn  = 16;
static constexpr int Cc  = 16;
static constexpr int Hh  = 192;
static constexpr int Ww  = 192;
static constexpr int HID = 128;

__device__ __forceinline__ float relu6f(float v) { return fminf(fmaxf(v, 0.f), 6.f); }

// Pack two f32 into one dword of bf16 pairs (lo, hi) -> v_cvt_pk_bf16_f32.
__device__ __forceinline__ unsigned pkbf(float lo, float hi) {
  unsigned short ul = __builtin_bit_cast(unsigned short, (__bf16)lo);
  unsigned short uh = __builtin_bit_cast(unsigned short, (__bf16)hi);
  return ((unsigned)uh << 16) | (unsigned)ul;
}

// dst[lane] = src[lane ^ 16] in one VALU op (v_permlanex16_b32, wave32).
__device__ __forceinline__ unsigned xhalf_u(unsigned u) {
  return __builtin_amdgcn_permlanex16(u, u, 0x76543210u, 0xFEDCBA98u, false, false);
}

// Load one 16x32 bf16 A-fragment from an LDS row-major [M][Kdim] weight matrix.
// Lane m=l%16 holds row m; lanes 0-15 hold K = kbase+{0..15},
// lanes 16-31 hold K = kbase+{16..31}. Two ds_load_b128 per lane.
__device__ __forceinline__ v16bf lds_afrag(const __bf16* W, int Kdim, int m, int q, int kbase) {
  const __bf16* p0 = W + m * Kdim + kbase + 16 * q;
  v8bf lo = *(const v8bf*)p0;
  v8bf hi = *(const v8bf*)(p0 + 8);
  v16bf a;
#pragma unroll
  for (int i = 0; i < 8; ++i) { a[i] = lo[i]; a[i + 8] = hi[i]; }
  return a;
}

// D-layout -> B-layout relayout for one K=32 fragment, on packed bf16 dwords.
// r0 = relu6(bias+acc[2f]) floats, r1 = same for acc[2f+1].
__device__ __forceinline__ v16bf relayout_frag(const float* r0, const float* r1, int q) {
  v8u bu;
#pragma unroll
  for (int i = 0; i < 4; ++i) {
    unsigned p0 = pkbf(r0[2 * i], r0[2 * i + 1]);
    unsigned p1 = pkbf(r1[2 * i], r1[2 * i + 1]);
    unsigned c0 = xhalf_u(p0);
    unsigned c1 = xhalf_u(p1);
    bu[i]     = q ? c1 : p0;   // elements 2i,2i+1   (K' = 16q + 2i,2i+1)
    bu[4 + i] = q ? p1 : c0;   // elements 8+2i,9+2i (K' = 16q + 8+2i,...)
  }
  return __builtin_bit_cast(v16bf, bu);
}

__global__ __launch_bounds__(256) void nca_mlp_kernel(
    const float* __restrict__ x,  const float* __restrict__ w1, const float* __restrict__ b1,
    const float* __restrict__ w2, const float* __restrict__ b2, const float* __restrict__ w3,
    const float* __restrict__ b3, const float* __restrict__ fire,
    float* __restrict__ out, float* __restrict__ ws_alpha)
{
  // Weights as bf16 in LDS. W1 padded to K=64 with column 48 = b1 (bias folded
  // via a constant-1 feature row), cols 49..63 = 0.
  __shared__ __bf16 sW1[HID * 64];
  __shared__ __bf16 sW2[HID * HID];
  __shared__ __bf16 sW3[16 * HID];
  __shared__ float  sB2[HID];
  __shared__ float  sB3[16];

  const int tid = threadIdx.x;
  for (int i = tid; i < HID * 64; i += 256) {
    int o = i >> 6, k = i & 63;
    float v = (k < 48) ? w1[o * 48 + k] : (k == 48 ? b1[o] : 0.f);
    sW1[i] = (__bf16)v;
  }
  for (int i = tid; i < HID * HID; i += 256) sW2[i] = (__bf16)w2[i];
  for (int i = tid; i < 16 * HID;  i += 256) sW3[i] = (__bf16)w3[i];
  if (tid < HID) sB2[tid] = b2[tid];
  if (tid < 16)  sB3[tid] = b3[tid];

  const int lane = tid & 31;
  const int n    = lane & 15;   // pixel / column index within 16-wide group
  const int q    = lane >> 4;   // lane half (0/1)
  const int NG   = Bn * Hh * (Ww / 16);
  const int gw0  = blockIdx.x * 8 + (tid >> 5);
  const int gstr = gridDim.x * 8;

  // NG / gstr is exactly 4 for every wave -> uniform trip count, barrier-safe.
  for (int g = gw0; g < NG; g += gstr) {
    // Barrier each iteration: acts as an LDS fence so the compiler cannot hoist
    // the weight-fragment ds_loads out of this loop (round-1 did, and spilled
    // 416 VGPRs of fragments to scratch).
    __syncthreads();

    int wg = g % (Ww / 16);
    int t0 = g / (Ww / 16);
    int h  = t0 % Hh;
    int b  = t0 / Hh;
    int wpix = wg * 16 + n;

    // ---- Perception: build B-fragments of y (64 x 16) in bf16 ----
    // frag0: K=0..15 -> x channels (lo lanes), K=16..31 -> sobel_x (hi lanes)
    // frag1: K=32..47 -> sobel_y (lo lanes), K=48 -> 1.0 (bias row), 49..63 -> 0 (hi lanes)
    v16bf by0, by1;
    const float* xb = x + (size_t)b * Cc * Hh * Ww;
#pragma unroll
    for (int c = 0; c < Cc; ++c) {
      const float* xp = xb + (size_t)c * Hh * Ww;
      float wv[3][3];
#pragma unroll
      for (int di = 0; di < 3; ++di) {
        int hh = h + di - 1;
        bool hok = (hh >= 0) && (hh < Hh);
#pragma unroll
        for (int dj = 0; dj < 3; ++dj) {
          int wp = wpix + dj - 1;
          bool ok = hok && (wp >= 0) && (wp < Ww);
          wv[di][dj] = ok ? xp[hh * Ww + wp] : 0.f;
        }
      }
      float gx  = (-wv[0][0] + wv[0][2] - 2.f * wv[1][0] + 2.f * wv[1][2] - wv[2][0] + wv[2][2]) * 0.125f;
      float gy  = (-wv[0][0] - 2.f * wv[0][1] - wv[0][2] + wv[2][0] + 2.f * wv[2][1] + wv[2][2]) * 0.125f;
      float ctr = wv[1][1];
      by0[c] = (__bf16)(q ? gx : ctr);
      by1[c] = (__bf16)(q ? (c == 0 ? 1.f : 0.f) : gy);
    }

    // ---- Layer 1: h1 = relu6(W1p @ y)  (bias folded into K=48 row) ----
    v8f acc[8];
#pragma unroll
    for (int tt = 0; tt < 8; ++tt) {
      v8f a{};
      a = __builtin_amdgcn_wmma_f32_16x16x32_bf16(false, lds_afrag(sW1, 64, tt * 16 + n, q, 0),
                                                  false, by0, (short)0, a, false, false);
      a = __builtin_amdgcn_wmma_f32_16x16x32_bf16(false, lds_afrag(sW1, 64, tt * 16 + n, q, 32),
                                                  false, by1, (short)0, a, false, false);
      acc[tt] = a;
    }

    v16bf bh[4];
#pragma unroll
    for (int f = 0; f < 4; ++f) {
      float r0[8], r1[8];
#pragma unroll
      for (int j = 0; j < 8; ++j) {
        r0[j] = relu6f(acc[2 * f][j]);
        r1[j] = relu6f(acc[2 * f + 1][j]);
      }
      bh[f] = relayout_frag(r0, r1, q);
    }

    // ---- Layer 2: h2 = relu6(W2 @ h1 + b2) ----
#pragma unroll
    for (int tt = 0; tt < 8; ++tt) {
      v8f a{};
#pragma unroll
      for (int k = 0; k < 4; ++k)
        a = __builtin_amdgcn_wmma_f32_16x16x32_bf16(false, lds_afrag(sW2, 128, tt * 16 + n, q, 32 * k),
                                                    false, bh[k], (short)0, a, false, false);
      acc[tt] = a;
    }
    v16bf bh2[4];
#pragma unroll
    for (int f = 0; f < 4; ++f) {
      // bias rows for acc[2f]: 32f+8q+{0..7}; for acc[2f+1]: 32f+16+8q+{0..7}
      v4f ba0 = *(const v4f*)&sB2[32 * f + 8 * q];
      v4f ba1 = *(const v4f*)&sB2[32 * f + 8 * q + 4];
      v4f bb0 = *(const v4f*)&sB2[32 * f + 16 + 8 * q];
      v4f bb1 = *(const v4f*)&sB2[32 * f + 16 + 8 * q + 4];
      float r0[8], r1[8];
#pragma unroll
      for (int j = 0; j < 8; ++j) {
        float biasa = (j < 4) ? ba0[j & 3] : ba1[j & 3];
        float biasb = (j < 4) ? bb0[j & 3] : bb1[j & 3];
        r0[j] = relu6f(acc[2 * f][j] + biasa);
        r1[j] = relu6f(acc[2 * f + 1][j] + biasb);
      }
      bh2[f] = relayout_frag(r0, r1, q);
    }

    // ---- Layer 3: dx = W3 @ h2 + b3 (single 16x16 tile) ----
    v8f d{};
#pragma unroll
    for (int k = 0; k < 4; ++k)
      d = __builtin_amdgcn_wmma_f32_16x16x32_bf16(false, lds_afrag(sW3, 128, n, q, 32 * k),
                                                  false, bh2[k], (short)0, d, false, false);

    // ---- Epilogue: x_new = x + dx * fire_mask ----
    v4f b3a = *(const v4f*)&sB3[8 * q];
    v4f b3b = *(const v4f*)&sB3[8 * q + 4];
    float fn = fire[((size_t)b * Hh + h) * Ww + wpix];
    float um = (fn <= 0.5f) ? 1.f : 0.f;
#pragma unroll
    for (int j = 0; j < 8; ++j) {
      int c = j + 8 * q;                      // D row = output channel
      float bias3 = (j < 4) ? b3a[j & 3] : b3b[j & 3];
      size_t idx = (((size_t)b * Cc + c) * Hh + h) * Ww + wpix;
      float xn = x[idx] + (d[j] + bias3) * um;
      out[idx] = xn;
      if (c == 3) ws_alpha[((size_t)b * Hh + h) * Ww + wpix] = xn;  // stage new alpha
    }
  }
}

// Pass 2: life mask = (3x3 max of old alpha > 0.1) & (3x3 max of new alpha > 0.1)
__global__ __launch_bounds__(256) void nca_mask_kernel(
    const float* __restrict__ x, const float* __restrict__ ws_alpha, float* __restrict__ out)
{
  int p = blockIdx.x * 256 + threadIdx.x;
  int total = Bn * Hh * Ww;
  if (p >= total) return;
  int w = p % Ww;
  int t = p / Ww;
  int h = t % Hh;
  int b = t / Hh;
  const float* xa = x + (((size_t)b * Cc + 3) * Hh) * Ww;  // old alpha plane
  const float* na = ws_alpha + (size_t)b * Hh * Ww;        // new alpha plane
  float mpre = -1e30f, mpost = -1e30f;
#pragma unroll
  for (int di = -1; di <= 1; ++di) {
    int hh = h + di;
    if (hh < 0 || hh >= Hh) continue;
#pragma unroll
    for (int dj = -1; dj <= 1; ++dj) {
      int ww2 = w + dj;
      if (ww2 < 0 || ww2 >= Ww) continue;
      mpre  = fmaxf(mpre,  xa[hh * Ww + ww2]);
      mpost = fmaxf(mpost, na[hh * Ww + ww2]);
    }
  }
  float alive = (mpre > 0.1f && mpost > 0.1f) ? 1.f : 0.f;
#pragma unroll
  for (int c = 0; c < Cc; ++c) {
    size_t idx = (((size_t)b * Cc + c) * Hh + h) * Ww + w;
    out[idx] *= alive;
  }
}

extern "C" void kernel_launch(void* const* d_in, const int* in_sizes, int n_in,
                              void* d_out, int out_size, void* d_ws, size_t ws_size,
                              hipStream_t stream)
{
  const float* x    = (const float*)d_in[0];
  const float* w1   = (const float*)d_in[1];
  const float* b1   = (const float*)d_in[2];
  const float* w2   = (const float*)d_in[3];
  const float* b2   = (const float*)d_in[4];
  const float* w3   = (const float*)d_in[5];
  const float* b3   = (const float*)d_in[6];
  const float* fire = (const float*)d_in[7];
  float* out = (float*)d_out;
  float* wsa = (float*)d_ws;   // needs Bn*Hh*Ww*4 = 2.25 MB for the new-alpha plane

  // 36864 wave-groups of 16 pixels; 8 waves/block, 1152 blocks -> exactly 4
  // groups/wave (uniform trip count for the in-loop barrier).
  nca_mlp_kernel<<<1152, 256, 0, stream>>>(x, w1, b1, w2, b2, w3, b3, fire, out, wsa);

  int total = Bn * Hh * Ww;
  nca_mask_kernel<<<(total + 255) / 256, 256, 0, stream>>>(x, wsa, out);
}
